// GATConv_10453950399052
// MI455X (gfx1250) — matrix operand order repaired and checked by reading, not compiled, not run on hardware
//
#include <hip/hip_runtime.h>

// ---------------------------------------------------------------------------
// GATConv, N=8192, F=512, H=1, C=128, dense-softmax rewrite:
//   out[i,:] = elu( (colsum(ft) + sum_{j:adj!=0} (e_ij-1)*ft[j,:]) / denom[i] )
//   denom[i] = N + sum_{j:adj!=0} (e_ij-1),  e_ij = exp(lrelu(a_src[i]+a_dst[j]))
// Roofline: one 256 MB stream of adjacency @ 23.3 TB/s ~= 11 us; all other
// work (1.07 GFLOP bf16-WMMA GEMM, 4 MB L2-resident ft, ~270K edges) is noise.
// ---------------------------------------------------------------------------

typedef __attribute__((ext_vector_type(16))) __bf16 v16bf;
typedef __attribute__((ext_vector_type(8)))  float  v8f;
typedef __attribute__((ext_vector_type(4)))  float  f4;

#define NN   8192
#define FF   512
#define CC   128
#define NEG_SLOPE 0.2f
#define CHUNK 1024   // adjacency columns per block iteration (4 KB)

static __device__ __forceinline__ unsigned short f2bf(float f) {
  union { float f; unsigned int u; } x; x.f = f;
  unsigned int r = x.u + 0x7FFFu + ((x.u >> 16) & 1u);   // round-to-nearest-even
  return (unsigned short)(r >> 16);
}
static __device__ __forceinline__ unsigned int f2bf2(float lo, float hi) {
  return (unsigned int)f2bf(lo) | ((unsigned int)f2bf(hi) << 16);
}

union AFrag { v16bf v; unsigned int u[8]; };

// ---------------------------------------------------------------------------
// Kernel 1: ft = X @ W   (8192x512 @ 512x128), bf16 WMMA, f32 accumulate.
// Block: 256 thr (8 waves). Block tile: 128(M) x 128(N). Wave tile: 16 x 128.
// ---------------------------------------------------------------------------
__global__ __launch_bounds__(256) void gemm_ft_wmma(
    const float* __restrict__ X, const float* __restrict__ W,
    float* __restrict__ FT)
{
  __shared__ unsigned short lA[128][32];   // bf16 X tile, row-major [m][k]
  __shared__ unsigned short lB[128][32];   // bf16 W tile, transposed [n][k]

  const int tid  = threadIdx.x;
  const int wave = tid >> 5;
  const int lane = tid & 31;
  const int blockRow = blockIdx.x * 128;

  v8f acc[8];
#pragma unroll
  for (int t = 0; t < 8; ++t) acc[t] = (v8f)(0.0f);

  const int ar = tid >> 1, ah = (tid & 1) * 16;     // A stage: 16 floats / thr
  const int br = tid >> 3, bc = (tid & 7) * 16;     // B stage: 16 floats / thr
  const int ml = wave * 16 + (lane & 15);
  const int kh = lane >> 4;

  for (int k0 = 0; k0 < FF; k0 += 32) {
    // --- stage A tile (128x32 f32 -> bf16 LDS), packed pair stores ---
    const float* xs = X + (size_t)(blockRow + ar) * FF + k0 + ah;
#pragma unroll
    for (int q = 0; q < 16; q += 4) {
      f4 v = *(const f4*)(xs + q);
      *(unsigned int*)&lA[ar][ah + q + 0] = f2bf2(v.x, v.y);
      *(unsigned int*)&lA[ar][ah + q + 2] = f2bf2(v.z, v.w);
    }
    // --- stage B tile (32x128 f32 -> transposed bf16 LDS [n][k]) ---
    const float* wsrc = W + (size_t)(k0 + br) * CC + bc;
#pragma unroll
    for (int q = 0; q < 16; q += 4) {
      f4 v = *(const f4*)(wsrc + q);
      lB[bc + q + 0][br] = f2bf(v.x);
      lB[bc + q + 1][br] = f2bf(v.y);
      lB[bc + q + 2][br] = f2bf(v.z);
      lB[bc + q + 3][br] = f2bf(v.w);
    }
    __syncthreads();

    // --- A fragment: lane -> row ml, K-half kh; elem e: K=(e>>3)*16+kh*8+(e&7)
    AFrag a;
#pragma unroll
    for (int p = 0; p < 4; ++p) {
      a.u[p]     = *(const unsigned int*)&lA[ml][kh * 8 + 2 * p];
      a.u[p + 4] = *(const unsigned int*)&lA[ml][16 + kh * 8 + 2 * p];
    }
    // --- 8 B fragments / 8 WMMAs ---
#pragma unroll
    for (int t = 0; t < 8; ++t) {
      const int n = t * 16 + (lane & 15);
      AFrag b;
#pragma unroll
      for (int p = 0; p < 8; ++p)
        b.u[p] = *(const unsigned int*)&lB[n][(p >> 2) * 16 + kh * 8 + 2 * (p & 3)];
      acc[t] = __builtin_amdgcn_wmma_f32_16x16x32_bf16(
          false, a.v, false, b.v, (short)0, acc[t], false, false);
    }
    __syncthreads();
  }

  // D layout: VGPR v -> M = v + 8*(lane>>4), N = lane&15 (+16*t)
  const int mbase = blockRow + wave * 16 + (lane >> 4) * 8;
  const int nbase = lane & 15;
#pragma unroll
  for (int t = 0; t < 8; ++t) {
#pragma unroll
    for (int v = 0; v < 8; ++v)
      FT[(size_t)(mbase + v) * CC + t * 16 + nbase] = acc[t][v];
  }
}

// ---------------------------------------------------------------------------
// Kernel 2: a_src[i] = <ft[i,:], att_src>, a_dst[i] = <ft[i,:], att_dst>
// One wave (32 lanes x 4ch) per row; 8 rows per 256-thr block. wave32 shfl.
// ---------------------------------------------------------------------------
__global__ __launch_bounds__(256) void att_scores(
    const float* __restrict__ FT, const float* __restrict__ att_src,
    const float* __restrict__ att_dst, float* __restrict__ asrc,
    float* __restrict__ adst)
{
  const int wave = threadIdx.x >> 5, lane = threadIdx.x & 31;
  const int row = blockIdx.x * 8 + wave;
  f4 v  = *(const f4*)(FT + (size_t)row * CC + lane * 4);
  f4 as = *(const f4*)(att_src + lane * 4);
  f4 ad = *(const f4*)(att_dst + lane * 4);
  float s = v.x * as.x + v.y * as.y + v.z * as.z + v.w * as.w;
  float d = v.x * ad.x + v.y * ad.y + v.z * ad.z + v.w * ad.w;
#pragma unroll
  for (int off = 16; off > 0; off >>= 1) {
    s += __shfl_xor(s, off, 32);
    d += __shfl_xor(d, off, 32);
  }
  if (lane == 0) { asrc[row] = s; adst[row] = d; }
}

// ---------------------------------------------------------------------------
// Kernel 3a/3b: S[c] = sum_i ft[i,c]
// ---------------------------------------------------------------------------
__global__ void zeroS(float* S) { S[threadIdx.x] = 0.0f; }

__global__ __launch_bounds__(256) void colsum_ft(
    const float* __restrict__ FT, float* __restrict__ S)
{
  __shared__ float tmp[256];
  const int tid = threadIdx.x;
  const int c = tid & 127, half = tid >> 7;
  const int r0 = blockIdx.x * 128;
  float s = 0.0f;
  for (int r = half; r < 128; r += 2) s += FT[(size_t)(r0 + r) * CC + c];
  tmp[tid] = s;
  __syncthreads();
  if (tid < 128) atomicAdd(&S[c], tmp[tid] + tmp[tid + 128]);
}

// ---------------------------------------------------------------------------
// Kernel 4: per-row streaming pass over adjacency (the 256 MB / 11 us floor).
// One 128-thread block per row; thread c owns output channel c.
// Non-temporal B128 loads (single-use data > L2) + global_prefetch of the
// next chunk so HBM requests stay outstanding across the compaction barrier.
// ---------------------------------------------------------------------------
__global__ __launch_bounds__(128) void gat_row(
    const float* __restrict__ adj, const float* __restrict__ FT,
    const float* __restrict__ asrc, const float* __restrict__ adst,
    const float* __restrict__ S, float* __restrict__ out)
{
  __shared__ int   cnt;
  __shared__ int   jl[CHUNK];
  __shared__ float wl[CHUNK];
  __shared__ float red[128];

  const int row = blockIdx.x;
  const int tid = threadIdx.x;
  const float ai = asrc[row];
  const float* __restrict__ arow = adj + (size_t)row * NN;

  float acc = 0.0f;     // sum (e-1)*ft[j, tid]
  float dloc = 0.0f;    // partial of sum (e-1)

  for (int c0 = 0; c0 < NN; c0 += CHUNK) {
    if (tid == 0) cnt = 0;

    const int j0 = c0 + tid * 8;
    f4 a0 = __builtin_nontemporal_load((const f4*)(arow + j0));
    f4 a1 = __builtin_nontemporal_load((const f4*)(arow + j0 + 4));

    // keep next chunk's lines in flight across the barrier (one prefetch
    // per 128B cacheline: 4 threads/line at 32B each)
    if (c0 + CHUNK < NN && (tid & 3) == 0)
      __builtin_prefetch(arow + j0 + CHUNK, 0, 0);

    __syncthreads();

#pragma unroll
    for (int q = 0; q < 8; ++q) {
      float av = (q < 4) ? a0[q & 3] : a1[q & 3];
      if (av != 0.0f) {
        const int j = j0 + q;
        float m = (ai + adst[j]) * av;               // scores * adjacency
        m = (m >= 0.0f) ? m : NEG_SLOPE * m;         // leaky relu
        float w = __expf(m) - 1.0f;                  // e_ij - 1
        int p = atomicAdd(&cnt, 1);                  // LDS compaction
        jl[p] = j; wl[p] = w;
        dloc += w;
      }
    }
    __syncthreads();

    const int n = cnt;
    for (int k = 0; k < n; ++k)
      acc += wl[k] * FT[(size_t)jl[k] * CC + tid];   // ft is L2-resident (4MB)
    __syncthreads();
  }

  // denom reduction across 128 channel-threads
  red[tid] = dloc;
  __syncthreads();
#pragma unroll
  for (int s2 = 64; s2 > 0; s2 >>= 1) {
    if (tid < s2) red[tid] += red[tid + s2];
    __syncthreads();
  }
  const float denom = (float)NN + red[0];

  float o = (S[tid] + acc) / denom;
  out[(size_t)row * CC + tid] = (o > 0.0f) ? o : (__expf(o) - 1.0f);
}

// ---------------------------------------------------------------------------
extern "C" void kernel_launch(void* const* d_in, const int* in_sizes, int n_in,
                              void* d_out, int out_size, void* d_ws, size_t ws_size,
                              hipStream_t stream)
{
  const float* features = (const float*)d_in[0];   // (8192, 512)
  const float* adjacency = (const float*)d_in[1];  // (8192, 8192)
  const float* kernel_w = (const float*)d_in[2];   // (512, 128)
  const float* att_src = (const float*)d_in[3];    // (1,1,128)
  const float* att_dst = (const float*)d_in[4];    // (1,1,128)
  float* out = (float*)d_out;                      // (8192, 128)

  float* FT   = (float*)d_ws;                      // 8192*128
  float* asrc = FT + (size_t)NN * CC;              // 8192
  float* adst = asrc + NN;                         // 8192
  float* S    = adst + NN;                         // 128

  gemm_ft_wmma<<<NN / 128, 256, 0, stream>>>(features, kernel_w, FT);
  att_scores  <<<NN / 8,   256, 0, stream>>>(FT, att_src, att_dst, asrc, adst);
  zeroS       <<<1,        128, 0, stream>>>(S);
  colsum_ft   <<<NN / 128, 256, 0, stream>>>(FT, S);
  gat_row     <<<NN,       128, 0, stream>>>(adjacency, FT, asrc, adst, S, out);
}
